// FusionGATTrackerV4_70385924047133
// MI455X (gfx1250) — matrix operand order repaired and checked by reading, not compiled, not run on hardware
//
#include <hip/hip_runtime.h>
#include <hip/hip_bf16.h>
#include <math.h>

#define CDIV(a,b) (((a)+(b)-1)/(b))

typedef __attribute__((ext_vector_type(16))) __bf16 v16bf;
typedef __attribute__((ext_vector_type(8)))  float  v8f;
typedef __attribute__((ext_vector_type(4)))  unsigned int v4u;
union BF16x16 { v4u u4[2]; v16bf v; unsigned short u[16]; };

__device__ __forceinline__ unsigned short f2bf(float f) {
  unsigned int u = __float_as_uint(f);
  unsigned int r = (u + 0x7FFFu + ((u >> 16) & 1u)) >> 16;
  return (unsigned short)r;
}

__device__ __forceinline__ void atomicMaxF(float* addr, float v) {
  if (v >= 0.f) atomicMax((int*)addr, __float_as_int(v));
  else          atomicMin((unsigned int*)addr, __float_as_uint(v));
}

// ---------------------------------------------------------------------------
// WMMA GEMM: Y[M x NCOLS] = act(Xbf[M x K] @ W[K x NCOLS] + bias)
// W pre-swizzled fragment-major (wswz_kernel); per block the 64-col weight
// slab is staged once in LDS, B-frags come from ds_load_b128.
// One wave computes 32 rows x 64 cols (2 A-frags share each B-frag).
// ACT: 0=none, 1=relu, 2=tanh.  BF: output bf16 (else f32).
// NCOLS compile-time -> branch-free epilogue, immediate-offset store clauses.
// ---------------------------------------------------------------------------
template<int K, int ACT, int NCOLS, bool BF>
__global__ __launch_bounds__(256) void wmma_gemm_kernel(
    const unsigned short* __restrict__ X, const unsigned short* __restrict__ Wsw,
    const float* __restrict__ bias, void* __restrict__ Yout, int M) {
  __shared__ __align__(16) unsigned short ldsW[4096];   // up to (64/32)*4*512
  const int lane = threadIdx.x & 31;
  const int wave = __builtin_amdgcn_readfirstlane(threadIdx.x >> 5);
  const int row0 = (blockIdx.x * 8 + wave) * 32;
  const int col0 = blockIdx.y * 64;

  // cooperative stage of this block's swizzled weight tiles (contiguous)
  const unsigned short* gW = Wsw + (size_t)(col0 >> 6) * ((K >> 5) * 2048);
  for (int i = threadIdx.x * 8; i < (K >> 5) * 2048; i += 256 * 8)
    *(v4u*)(ldsW + i) = *(const v4u*)(gW + i);
  __syncthreads();
  if (row0 >= M) return;                        // wave-uniform exit after barrier

  v8f z = {0.f,0.f,0.f,0.f,0.f,0.f,0.f,0.f};
  v8f acc[2][4] = {{z,z,z,z},{z,z,z,z}};

  const int m   = lane & 15;                    // A-frag row
  const int kb8 = (lane >> 4) * 16;             // A-frag K base (0/16)

#pragma unroll
  for (int ks = 0; ks < K; ks += 32) {
    BF16x16 a[2];
#pragma unroll
    for (int h = 0; h < 2; ++h) {
      const unsigned short* arow = X + (size_t)(row0 + h * 16 + m) * K + ks;
      a[h].u4[0] = *(const v4u*)(arow + kb8);
      a[h].u4[1] = *(const v4u*)(arow + kb8 + 8);
    }
#pragma unroll
    for (int nt = 0; nt < 4; ++nt) {
      BF16x16 b;
      const unsigned short* wp = ldsW + (ks >> 5) * 2048 + nt * 512 + lane * 16;
      b.u4[0] = *(const v4u*)(wp);
      b.u4[1] = *(const v4u*)(wp + 8);
      acc[0][nt] = __builtin_amdgcn_wmma_f32_16x16x32_bf16(
          false, a[0].v, false, b.v, (short)0, acc[0][nt], false, false);
      acc[1][nt] = __builtin_amdgcn_wmma_f32_16x16x32_bf16(
          false, a[1].v, false, b.v, (short)0, acc[1][nt], false, false);
    }
  }

  const int n16 = lane & 15;
  const int mh  = (lane >> 4) * 8;
  float bv[4];
#pragma unroll
  for (int nt = 0; nt < 4; ++nt) bv[nt] = bias[col0 + nt * 16 + n16];
#pragma unroll
  for (int h = 0; h < 2; ++h) {
#pragma unroll
    for (int nt = 0; nt < 4; ++nt) {
      const size_t basei = (size_t)(row0 + h * 16 + mh) * NCOLS + col0 + nt * 16 + n16;
#pragma unroll
      for (int r = 0; r < 8; ++r) {
        float v = acc[h][nt][r] + bv[nt];
        if (ACT == 1) v = fmaxf(v, 0.f);
        if (ACT == 2) v = tanhf(v);
        if (BF) ((unsigned short*)Yout)[basei + (size_t)r * NCOLS] = f2bf(v);
        else    ((float*)Yout)[basei + (size_t)r * NCOLS] = v;
      }
    }
  }
}

// ---------------------------------------------------------------------------
// Weight swizzle: src f32 [K0 x Ncols] (or [Ncols x Kp] if transposed) ->
// dst bf16 fragment-major: idx = ((colBlk*(Kp/32)+ksBlk)*4 + nt)*512 + lane*16 + j
// where lane 0-15 -> K half 0, lane 16-31 -> K half 1; rows [K0,Kp) zero-padded.
// ---------------------------------------------------------------------------
__global__ void wswz_kernel(const float* __restrict__ src, unsigned short* __restrict__ dst,
                            int K0, int Kp, int Ncols, int transposed) {
  int idx = blockIdx.x * blockDim.x + threadIdx.x;
  if (idx >= Kp * Ncols) return;
  int j    = idx & 15;
  int lane = (idx >> 4) & 31;
  int tile = idx >> 9;
  int nt   = tile & 3;
  int kBlocks = Kp >> 5;
  int ksBlk = (tile >> 2) % kBlocks;
  int colBlk = tile / (4 * kBlocks);
  int k   = ksBlk * 32 + ((lane >> 4) << 4) + j;
  int col = colBlk * 64 + nt * 16 + (lane & 15);
  float v = 0.f;
  if (k < K0) v = transposed ? src[(size_t)col * Kp + k] : src[(size_t)k * Ncols + col];
  dst[idx] = f2bf(v);
}

// ---------------------------------------------------------------------------
// Small helper kernels
// ---------------------------------------------------------------------------
__global__ void fill_kernel(float* p, float v, int n) {
  int i = blockIdx.x * blockDim.x + threadIdx.x;
  if (i < n) p[i] = v;
}

__global__ void tobf_kernel(const float* __restrict__ src, unsigned short* __restrict__ dst, int n) {
  int i = blockIdx.x * blockDim.x + threadIdx.x;
  if (i < n) dst[i] = f2bf(src[i]);
}

// node_feat = concat(x[7], pos[3], type_emb[node_type][8], zeros) -> bf16 [N x 32]
__global__ void build_nf_kernel(const float* __restrict__ x, const float* __restrict__ pos,
                                const int* __restrict__ ntyp, const float* __restrict__ temb,
                                unsigned short* __restrict__ nfbf, int N) {
  int idx = blockIdx.x * blockDim.x + threadIdx.x;
  if (idx >= N * 32) return;
  int i = idx >> 5, j = idx & 31;
  float v = 0.f;
  if (j < 7)       v = x[i * 7 + j];
  else if (j < 10) v = pos[i * 3 + (j - 7)];
  else if (j < 18) v = temb[ntyp[i] * 8 + (j - 10)];
  nfbf[idx] = f2bf(v);
}

__global__ void bias_act_kernel(const float* __restrict__ agg, const float* __restrict__ bias,
                                float* __restrict__ Yf, unsigned short* __restrict__ Ybf,
                                int total, int relu) {
  int idx = blockIdx.x * blockDim.x + threadIdx.x;
  if (idx >= total) return;
  float v = agg[idx] + bias[idx & 63];
  if (relu) v = fmaxf(v, 0.f);
  if (Yf) Yf[idx] = v;
  Ybf[idx] = f2bf(v);
}

// ---------------------------------------------------------------------------
// GATv2 edge kernels. One wave per edge (edge id forced scalar -> SMEM loads);
// 2 channels per lane. xe recomputed on the fly (6 FMAs/ch beats 512MB traffic).
// ---------------------------------------------------------------------------
__global__ __launch_bounds__(256) void gat_edge_logits_kernel(
    const int* __restrict__ ei, const float* __restrict__ ea,
    const float* __restrict__ We, const float* __restrict__ be,
    const float* __restrict__ xl, const float* __restrict__ xr,
    const float* __restrict__ att, float* __restrict__ logits,
    float* __restrict__ mmax, int E) {
  const int lane = threadIdx.x & 31;
  const int e = __builtin_amdgcn_readfirstlane(blockIdx.x * 8 + (threadIdx.x >> 5));
  if (e >= E) return;
  const int s = ei[e], d = ei[E + e];
  const float a0 = ea[e * 6 + 0], a1 = ea[e * 6 + 1], a2 = ea[e * 6 + 2];
  const float a3 = ea[e * 6 + 3], a4 = ea[e * 6 + 4], a5 = ea[e * 6 + 5];
  float part[2];
#pragma unroll
  for (int half = 0; half < 2; ++half) {
    const int c = lane + half * 32;
    float xe = be[c] + a0 * We[c] + a1 * We[64 + c] + a2 * We[128 + c]
             + a3 * We[192 + c] + a4 * We[256 + c] + a5 * We[320 + c];
    float v = xl[(size_t)s * 64 + c] + xr[(size_t)d * 64 + c] + xe;
    v = (v > 0.f) ? v : 0.2f * v;              // leaky_relu(0.2)
    part[half] = v * att[c];
  }
#pragma unroll
  for (int off = 8; off >= 1; off >>= 1) {     // reduce 16-lane head groups
    part[0] += __shfl_xor(part[0], off, 32);
    part[1] += __shfl_xor(part[1], off, 32);
  }
  if ((lane & 15) == 0) {
    const int hh = lane >> 4;                  // heads hh and hh+2
    logits[(size_t)e * 4 + hh]     = part[0];
    logits[(size_t)e * 4 + hh + 2] = part[1];
    atomicMaxF(&mmax[(size_t)d * 4 + hh],     part[0]);
    atomicMaxF(&mmax[(size_t)d * 4 + hh + 2], part[1]);
  }
}

__global__ void gat_edge_exp_kernel(const int* __restrict__ ei, const float* __restrict__ logits,
                                    const float* __restrict__ mmax, float* __restrict__ ssum,
                                    float* __restrict__ aout, int E) {
  int idx = blockIdx.x * blockDim.x + threadIdx.x;
  if (idx >= E * 4) return;
  int e = idx >> 2, h = idx & 3;
  int d = ei[E + e];
  float ex = __expf(logits[idx] - mmax[(size_t)d * 4 + h]);
  atomicAdd(&ssum[(size_t)d * 4 + h], ex);
  aout[idx] = ex;                              // alpha region temporarily holds ex
}

__global__ __launch_bounds__(256) void gat_edge_scatter_kernel(
    const int* __restrict__ ei, const float* __restrict__ xl,
    const float* __restrict__ ssum, float* __restrict__ aout,
    float* __restrict__ agg, int E) {
  const int lane = threadIdx.x & 31;
  const int e = __builtin_amdgcn_readfirstlane(blockIdx.x * 8 + (threadIdx.x >> 5));
  if (e >= E) return;
  const int s = ei[e], d = ei[E + e];
  float alpha[4];
#pragma unroll
  for (int h = 0; h < 4; ++h)
    alpha[h] = aout[(size_t)e * 4 + h] / (ssum[(size_t)d * 4 + h] + 1e-16f);
#pragma unroll
  for (int half = 0; half < 2; ++half) {
    const int c = lane + half * 32;
    atomicAdd(&agg[(size_t)d * 64 + c], xl[(size_t)s * 64 + c] * alpha[c >> 4]);
  }
  if (lane < 4) aout[(size_t)e * 4 + lane] = alpha[lane];   // finalize alpha
}

// ---------------------------------------------------------------------------
// Fusion-attention / decode / GRU small kernels
// ---------------------------------------------------------------------------
__global__ __launch_bounds__(256) void rowdot_kernel(
    const float* __restrict__ T1, const float* __restrict__ w,
    const float* __restrict__ b, float* __restrict__ out, int M) {
  const int lane = threadIdx.x & 31;
  const int i = __builtin_amdgcn_readfirstlane(blockIdx.x * 8 + (threadIdx.x >> 5));
  if (i >= M) return;
  float p = T1[(size_t)i * 64 + lane] * w[lane] + T1[(size_t)i * 64 + lane + 32] * w[lane + 32];
#pragma unroll
  for (int off = 16; off >= 1; off >>= 1) p += __shfl_xor(p, off, 32);
  if (lane == 0) out[i] = p + b[0];
}

__global__ void seg_max_kernel(const int* __restrict__ seg, const float* __restrict__ v,
                               float* __restrict__ cm, int M) {
  int i = blockIdx.x * blockDim.x + threadIdx.x;
  if (i < M) atomicMaxF(&cm[seg[i]], v[i]);
}

__global__ void seg_exp_kernel(const int* __restrict__ seg, const float* __restrict__ v,
                               const float* __restrict__ cm, float* __restrict__ cs,
                               float* __restrict__ wfa, int M) {
  int i = blockIdx.x * blockDim.x + threadIdx.x;
  if (i >= M) return;
  float ex = __expf(v[i] - cm[seg[i]]);
  atomicAdd(&cs[seg[i]], ex);
  wfa[i] = ex;
}

__global__ void seg_fuse_kernel(const int* __restrict__ seg, const float* __restrict__ wfa,
                                const float* __restrict__ cs, const float* __restrict__ meas,
                                float* __restrict__ fused, float* __restrict__ counts, int M) {
  int idx = blockIdx.x * blockDim.x + threadIdx.x;
  if (idx >= M * 64) return;
  int i = idx >> 6, c = idx & 63;
  int g = seg[i];
  float w = wfa[i] / (cs[g] + 1e-16f);
  atomicAdd(&fused[(size_t)g * 64 + c], meas[(size_t)i * 64 + c] * w);
  if (c == 0) atomicAdd(&counts[g], 1.0f);
}

__global__ void dec2_kernel(const float* __restrict__ D1, const float* __restrict__ W2,
                            const float* __restrict__ b2, const float* __restrict__ counts,
                            float* __restrict__ out, int M) {
  int idx = blockIdx.x * blockDim.x + threadIdx.x;
  if (idx >= M * 7) return;
  int i = idx / 7, j = idx % 7;
  float acc = b2[j];
#pragma unroll
  for (int c = 0; c < 64; ++c) acc += D1[(size_t)i * 64 + c] * W2[c * 7 + j];
  if (counts && !(counts[i] > 0.f)) acc = 0.f;
  out[idx] = acc;
}

__global__ void gru_kernel(const float* __restrict__ gi, const float* __restrict__ gh,
                           const float* __restrict__ hprev, float* __restrict__ nh_out,
                           unsigned short* __restrict__ nhbf, int Ntr) {
  int idx = blockIdx.x * blockDim.x + threadIdx.x;
  if (idx >= Ntr * 64) return;
  int i = idx >> 6, c = idx & 63;
  float ir = gi[(size_t)i * 192 + c],       hr = gh[(size_t)i * 192 + c];
  float iz = gi[(size_t)i * 192 + 64 + c],  hz = gh[(size_t)i * 192 + 64 + c];
  float in = gi[(size_t)i * 192 + 128 + c], hn = gh[(size_t)i * 192 + 128 + c];
  float r = 1.f / (1.f + __expf(-(ir + hr)));
  float z = 1.f / (1.f + __expf(-(iz + hz)));
  float n = tanhf(in + r * hn);
  float nh = (1.f - z) * n + z * hprev[idx];
  nh_out[idx] = nh;
  nhbf[idx] = f2bf(nh);
}

// ---------------------------------------------------------------------------
extern "C" void kernel_launch(void* const* d_in, const int* in_sizes, int n_in,
                              void* d_out, int out_size, void* d_ws, size_t ws_size,
                              hipStream_t stream) {
  if (n_in < 42) return;
  const float* x        = (const float*)d_in[0];
  const float* pos      = (const float*)d_in[1];
  const int*   ntyp     = (const int*)d_in[2];
  const int*   ei       = (const int*)d_in[3];
  const float* ea       = (const float*)d_in[4];
  const int*   mbatch   = (const int*)d_in[5];
  const float* hidden   = (const float*)d_in[8];
  const float* type_emb = (const float*)d_in[9];
  const float* enc_W1 = (const float*)d_in[10]; const float* enc_b1 = (const float*)d_in[11];
  const float* enc_W2 = (const float*)d_in[12]; const float* enc_b2 = (const float*)d_in[13];
  const float* fa_W1  = (const float*)d_in[14]; const float* fa_b1  = (const float*)d_in[15];
  const float* fa_W2  = (const float*)d_in[16]; const float* fa_b2  = (const float*)d_in[17];
  const float* dec_W1 = (const float*)d_in[18]; const float* dec_b1 = (const float*)d_in[19];
  const float* dec_W2 = (const float*)d_in[20]; const float* dec_b2 = (const float*)d_in[21];
  const float* gru_Wih = (const float*)d_in[22]; const float* gru_Whh = (const float*)d_in[23];
  const float* gru_bih = (const float*)d_in[24]; const float* gru_bhh = (const float*)d_in[25];
  const float *gWl[2], *gbl[2], *gWr[2], *gbr[2], *gWe[2], *gbe[2], *gatt[2], *gbias[2];
  for (int L = 0; L < 2; ++L) {
    int b = 26 + L * 8;
    gWl[L] = (const float*)d_in[b + 0]; gbl[L] = (const float*)d_in[b + 1];
    gWr[L] = (const float*)d_in[b + 2]; gbr[L] = (const float*)d_in[b + 3];
    gWe[L] = (const float*)d_in[b + 4]; gbe[L] = (const float*)d_in[b + 5];
    gatt[L] = (const float*)d_in[b + 6]; gbias[L] = (const float*)d_in[b + 7];
  }

  // host-derivable dims (no D2H reads of scalar inputs -> graph-capture safe)
  const int N   = in_sizes[0] / 7;
  const int E   = in_sizes[3] / 2;
  const int Nme = in_sizes[5];
  const int Ntr = in_sizes[8] / 64;
  const int C   = (out_size - Ntr * 64 - 8 * E) / 7 - Ntr;

  float* out = (float*)d_out;
  float* out_main = out;                                 // (C+Ntr) x 7
  float* out_nh   = out + (size_t)(C + Ntr) * 7;         // Ntr x 64
  float* out_a1   = out_nh + (size_t)Ntr * 64;           // E x 4
  float* out_a2   = out_a1 + (size_t)E * 4;              // E x 4

  // ---- carve workspace --------------------------------------------------
  char* base = (char*)d_ws;
  size_t off = 0;
  auto alloc = [&](size_t bytes) -> void* {
    void* p = base + off;
    off = (off + bytes + 255) & ~(size_t)255;
    return p;
  };
  unsigned short* wbEnc1 = (unsigned short*)alloc(32 * 64 * 2);
  unsigned short* wbEnc2 = (unsigned short*)alloc(64 * 64 * 2);
  unsigned short* wbWl[2], *wbWr[2];
  for (int L = 0; L < 2; ++L) {
    wbWl[L] = (unsigned short*)alloc(64 * 64 * 2);
    wbWr[L] = (unsigned short*)alloc(64 * 64 * 2);
  }
  unsigned short* wbFa1  = (unsigned short*)alloc(64 * 64 * 2);
  unsigned short* wbDec1 = (unsigned short*)alloc(64 * 64 * 2);
  unsigned short* wbIhT  = (unsigned short*)alloc(64 * 192 * 2);
  unsigned short* wbHhT  = (unsigned short*)alloc(64 * 192 * 2);
  unsigned short* nfbf   = (unsigned short*)alloc((size_t)N * 32 * 2);
  unsigned short* tbf    = (unsigned short*)alloc((size_t)N * 64 * 2);
  unsigned short* hbf    = (unsigned short*)alloc((size_t)N * 64 * 2);
  unsigned short* h1bf   = (unsigned short*)alloc((size_t)N * 64 * 2);
  unsigned short* h2bf   = (unsigned short*)alloc((size_t)N * 64 * 2);
  unsigned short* fusedbf= (unsigned short*)alloc((size_t)C * 64 * 2);
  unsigned short* nhbf   = (unsigned short*)alloc((size_t)Ntr * 64 * 2);
  unsigned short* hidbf  = (unsigned short*)alloc((size_t)Ntr * 64 * 2);
  float* xl     = (float*)alloc((size_t)N * 64 * 4);
  float* xr     = (float*)alloc((size_t)N * 64 * 4);
  float* agg    = (float*)alloc((size_t)N * 64 * 4);
  float* h2f    = (float*)alloc((size_t)N * 64 * 4);
  float* logits = (float*)alloc((size_t)E * 4 * 4);
  float* mmax   = (float*)alloc((size_t)N * 4 * 4);
  float* ssum   = (float*)alloc((size_t)N * 4 * 4);
  float* falog  = (float*)alloc((size_t)Nme * 4);
  float* wfa    = (float*)alloc((size_t)Nme * 4);
  float* cm     = (float*)alloc((size_t)C * 4);
  float* cs     = (float*)alloc((size_t)C * 4);
  float* counts = (float*)alloc((size_t)C * 4);
  float* fused  = (float*)alloc((size_t)C * 64 * 4);
  float* gi     = (float*)alloc((size_t)Ntr * 192 * 4);
  float* gh     = (float*)alloc((size_t)Ntr * 192 * 4);
  float* t1 = xl;   // reuse (free after GAT layers): fusion tanh activations
  float* d1 = xr;   // reuse: decoder hidden

  const int B = 256;
  const float NEG = -3.0e38f;

  // ---- weight prep: fp32 -> bf16 fragment-major swizzle ------------------
  hipLaunchKernelGGL(wswz_kernel, dim3(CDIV(32 * 64, B)), dim3(B), 0, stream, enc_W1, wbEnc1, 18, 32, 64, 0);
  hipLaunchKernelGGL(wswz_kernel, dim3(CDIV(64 * 64, B)), dim3(B), 0, stream, enc_W2, wbEnc2, 64, 64, 64, 0);
  for (int L = 0; L < 2; ++L) {
    hipLaunchKernelGGL(wswz_kernel, dim3(CDIV(64 * 64, B)), dim3(B), 0, stream, gWl[L], wbWl[L], 64, 64, 64, 0);
    hipLaunchKernelGGL(wswz_kernel, dim3(CDIV(64 * 64, B)), dim3(B), 0, stream, gWr[L], wbWr[L], 64, 64, 64, 0);
  }
  hipLaunchKernelGGL(wswz_kernel, dim3(CDIV(64 * 64, B)), dim3(B), 0, stream, fa_W1, wbFa1, 64, 64, 64, 0);
  hipLaunchKernelGGL(wswz_kernel, dim3(CDIV(64 * 64, B)), dim3(B), 0, stream, dec_W1, wbDec1, 64, 64, 64, 0);
  hipLaunchKernelGGL(wswz_kernel, dim3(CDIV(64 * 192, B)), dim3(B), 0, stream, gru_Wih, wbIhT, 64, 64, 192, 1);
  hipLaunchKernelGGL(wswz_kernel, dim3(CDIV(64 * 192, B)), dim3(B), 0, stream, gru_Whh, wbHhT, 64, 64, 192, 1);

  // ---- encoder: h = relu(nf@W1+b1)@W2+b2 (WMMA, bf16) -------------------
  hipLaunchKernelGGL(build_nf_kernel, dim3(CDIV(N * 32, B)), dim3(B), 0, stream, x, pos, ntyp, type_emb, nfbf, N);
  hipLaunchKernelGGL((wmma_gemm_kernel<32,1,64,true>), dim3(CDIV(N, 256), 1), dim3(B), 0, stream,
                     nfbf, wbEnc1, enc_b1, tbf, N);
  hipLaunchKernelGGL((wmma_gemm_kernel<64,0,64,true>), dim3(CDIV(N, 256), 1), dim3(B), 0, stream,
                     tbf, wbEnc2, enc_b2, hbf, N);

  // ---- two GATv2 layers -------------------------------------------------
  for (int L = 0; L < 2; ++L) {
    const unsigned short* Xin = (L == 0) ? hbf : h1bf;
    float* aout = (L == 0) ? out_a1 : out_a2;
    hipLaunchKernelGGL(fill_kernel, dim3(CDIV(N * 4, B)), dim3(B), 0, stream, mmax, NEG, N * 4);
    hipLaunchKernelGGL(fill_kernel, dim3(CDIV(N * 4, B)), dim3(B), 0, stream, ssum, 0.f, N * 4);
    hipLaunchKernelGGL(fill_kernel, dim3(CDIV(N * 64, B)), dim3(B), 0, stream, agg, 0.f, N * 64);
    hipLaunchKernelGGL((wmma_gemm_kernel<64,0,64,false>), dim3(CDIV(N, 256), 1), dim3(B), 0, stream,
                       Xin, wbWl[L], gbl[L], xl, N);
    hipLaunchKernelGGL((wmma_gemm_kernel<64,0,64,false>), dim3(CDIV(N, 256), 1), dim3(B), 0, stream,
                       Xin, wbWr[L], gbr[L], xr, N);
    hipLaunchKernelGGL(gat_edge_logits_kernel, dim3(CDIV(E, 8)), dim3(B), 0, stream,
                       ei, ea, gWe[L], gbe[L], xl, xr, gatt[L], logits, mmax, E);
    hipLaunchKernelGGL(gat_edge_exp_kernel, dim3(CDIV(E * 4, B)), dim3(B), 0, stream,
                       ei, logits, mmax, ssum, aout, E);
    hipLaunchKernelGGL(gat_edge_scatter_kernel, dim3(CDIV(E, 8)), dim3(B), 0, stream,
                       ei, xl, ssum, aout, agg, E);
    if (L == 0)
      hipLaunchKernelGGL(bias_act_kernel, dim3(CDIV(N * 64, B)), dim3(B), 0, stream,
                         agg, gbias[L], (float*)nullptr, h1bf, N * 64, 1);
    else
      hipLaunchKernelGGL(bias_act_kernel, dim3(CDIV(N * 64, B)), dim3(B), 0, stream,
                         agg, gbias[L], h2f, h2bf, N * 64, 0);
  }

  // ---- fusion attention over measurement nodes --------------------------
  hipLaunchKernelGGL((wmma_gemm_kernel<64,2,64,false>), dim3(CDIV(Nme, 256), 1), dim3(B), 0, stream,
                     h2bf, wbFa1, fa_b1, t1, Nme);
  hipLaunchKernelGGL(rowdot_kernel, dim3(CDIV(Nme, 8)), dim3(B), 0, stream, t1, fa_W2, fa_b2, falog, Nme);
  hipLaunchKernelGGL(fill_kernel, dim3(CDIV(C, B)), dim3(B), 0, stream, cm, NEG, C);
  hipLaunchKernelGGL(fill_kernel, dim3(CDIV(C, B)), dim3(B), 0, stream, cs, 0.f, C);
  hipLaunchKernelGGL(fill_kernel, dim3(CDIV(C, B)), dim3(B), 0, stream, counts, 0.f, C);
  hipLaunchKernelGGL(fill_kernel, dim3(CDIV(C * 64, B)), dim3(B), 0, stream, fused, 0.f, C * 64);
  hipLaunchKernelGGL(seg_max_kernel, dim3(CDIV(Nme, B)), dim3(B), 0, stream, mbatch, falog, cm, Nme);
  hipLaunchKernelGGL(seg_exp_kernel, dim3(CDIV(Nme, B)), dim3(B), 0, stream, mbatch, falog, cm, cs, wfa, Nme);
  hipLaunchKernelGGL(seg_fuse_kernel, dim3(CDIV(Nme * 64, B)), dim3(B), 0, stream,
                     mbatch, wfa, cs, h2f, fused, counts, Nme);

  // ---- decode clusters ---------------------------------------------------
  hipLaunchKernelGGL(tobf_kernel, dim3(CDIV(C * 64, B)), dim3(B), 0, stream, fused, fusedbf, C * 64);
  hipLaunchKernelGGL((wmma_gemm_kernel<64,1,64,false>), dim3(CDIV(C, 256), 1), dim3(B), 0, stream,
                     fusedbf, wbDec1, dec_b1, d1, C);
  hipLaunchKernelGGL(dec2_kernel, dim3(CDIV(C * 7, B)), dim3(B), 0, stream,
                     d1, dec_W2, dec_b2, counts, out_main, C);

  // ---- GRU over track nodes ---------------------------------------------
  hipLaunchKernelGGL(tobf_kernel, dim3(CDIV(Ntr * 64, B)), dim3(B), 0, stream, hidden, hidbf, Ntr * 64);
  hipLaunchKernelGGL((wmma_gemm_kernel<64,0,192,false>), dim3(CDIV(Ntr, 256), 3), dim3(B), 0, stream,
                     h2bf + (size_t)Nme * 64, wbIhT, gru_bih, gi, Ntr);
  hipLaunchKernelGGL((wmma_gemm_kernel<64,0,192,false>), dim3(CDIV(Ntr, 256), 3), dim3(B), 0, stream,
                     hidbf, wbHhT, gru_bhh, gh, Ntr);
  hipLaunchKernelGGL(gru_kernel, dim3(CDIV(Ntr * 64, B)), dim3(B), 0, stream,
                     gi, gh, hidden, out_nh, nhbf, Ntr);

  // ---- decode tracks ------------------------------------------------------
  hipLaunchKernelGGL((wmma_gemm_kernel<64,1,64,false>), dim3(CDIV(Ntr, 256), 1), dim3(B), 0, stream,
                     nhbf, wbDec1, dec_b1, d1, Ntr);
  hipLaunchKernelGGL(dec2_kernel, dim3(CDIV(Ntr * 7, B)), dim3(B), 0, stream,
                     d1, dec_W2, dec_b2, (const float*)nullptr, out_main + (size_t)C * 7, Ntr);

  (void)ws_size; (void)n_in;
}
